// GCN_d_10402410791109
// MI455X (gfx1250) — matrix operand order repaired and checked by reading, not compiled, not run on hardware
//
#include <hip/hip_runtime.h>

#define NROWS 10000
#define LDT   10048   // padded contraction length (multiple of 64) for transposed B buffers

typedef __attribute__((ext_vector_type(16))) __bf16   bf16x16;
typedef __attribute__((ext_vector_type(2)))  __bf16   bf16x2;
typedef __attribute__((ext_vector_type(8)))  float    f32x8;
typedef __attribute__((ext_vector_type(4)))  float    f32x4;
typedef __attribute__((ext_vector_type(4)))  unsigned u32x4;

union OpU { u32x4 q[2]; bf16x16 m; };

// f32 -> bf16 pair, packed into one dword (v_cvt_pk_bf16_f32 when available)
__device__ __forceinline__ unsigned pack2(float lo, float hi) {
#if __has_builtin(__builtin_amdgcn_cvt_pk_bf16_f32)
  bf16x2 p = __builtin_amdgcn_cvt_pk_bf16_f32(lo, hi);
  return __builtin_bit_cast(unsigned, p);
#else
  unsigned ul = __builtin_bit_cast(unsigned, lo);
  ul += 0x7FFFu + ((ul >> 16) & 1u);
  unsigned uh = __builtin_bit_cast(unsigned, hi);
  uh += 0x7FFFu + ((uh >> 16) & 1u);
  return (ul >> 16) | (uh & 0xFFFF0000u);
#endif
}
__device__ __forceinline__ unsigned short f2bf(float f) {
  return (unsigned short)(pack2(f, 0.0f) & 0xFFFFu);
}
__device__ __forceinline__ float bf2f(unsigned short h) {
  return __builtin_bit_cast(float, (unsigned)h << 16);
}

// ---------------------------------------------------------------------------
// Tiny projection GEMM: out^T[j][i] = bf16( sum_k X[i][k] * W[k][j] )
// Lane mapping: j varies across lanes -> X[i][k] is wave-uniform (scalar
// broadcast), W[k][j] is lane-coalesced. Transposed bf16 store (ld = LDT)
// zero-pads i in [NROWS, LDT) and j in [KoutValid, Kout) so the big GEMM
// needs no B-side guards.
// ---------------------------------------------------------------------------
__global__ __launch_bounds__(256) void xw_gemm_T(
    const float* __restrict__ Xf, const unsigned short* __restrict__ Xb,
    const float* __restrict__ W, int K, int Kout, int KoutValid,
    unsigned short* __restrict__ outT)
{
  const int idx = blockIdx.x * 256 + threadIdx.x;
  const int j = idx % Kout;                 // lane-contiguous
  const int i = idx / Kout;                 // wave-uniform (Kout multiple of 128)
  if (i >= LDT) return;
  float s = 0.0f;
  if (i < NROWS && j < KoutValid) {
    if (Xf) {
      for (int k = 0; k < K; ++k)
        s += Xf[(size_t)i * K + k] * W[(size_t)k * KoutValid + j];
    } else {
      for (int k = 0; k < K; ++k)
        s += bf2f(Xb[(size_t)i * K + k]) * W[(size_t)k * KoutValid + j];
    }
  }
  outT[(size_t)j * LDT + i] = f2bf(s);
}

// ---------------------------------------------------------------------------
// C = act(adj @ X):  adj [NROWS x NROWS] f32 streamed and converted to bf16
// into double-buffered LDS in WMMA A-register layout (BK=64 per barrier).
// XT = X^T [Kout x LDT] bf16, L2-resident, two 16B loads per lane (k 8h.., 16+8h..).
// WG tile 128x64, 8 waves as 4x2, 2x2 WMMA tiles per wave, 8 WMMA/barrier.
// ---------------------------------------------------------------------------
__global__ __launch_bounds__(256) void adj_gemm(
    const float* __restrict__ adj,
    const unsigned short* __restrict__ XT,
    unsigned short* __restrict__ outB, int ldB,
    float* __restrict__ outF, int ldF, int ncolF,
    int act)
{
  __shared__ unsigned lds[2][4096];          // 2 x 16KB: 128x64 bf16 per stage
  const int t    = threadIdx.x;
  const int lane = t & 31;
  const int wave = t >> 5;
  const int wr   = wave >> 1, wc = wave & 1;
  const int rowBase = blockIdx.x * 128;
  const int colBase = blockIdx.y * 64;

  // Staging assignment: thread covers row r, 32 consecutive k starting at c0.
  // Rows >= NROWS are clamped to row NROWS-1 (their outputs are discarded),
  // so the interior loop has no control flow at all.
  const int r    = t >> 1;
  const int c0   = (t & 1) * 32;
  const int grow = rowBase + r;
  const float* __restrict__ arow =
      adj + (size_t)(grow < NROWS ? grow : NROWS - 1) * NROWS;

  f32x8 acc[2][2] = {};

  auto consume = [&](const unsigned* stage, int k0) {
    #pragma unroll
    for (int kh = 0; kh < 2; ++kh) {
      OpU a[2], b[2];
      #pragma unroll
      for (int tr = 0; tr < 2; ++tr) {
        const unsigned* ap = stage + kh * 2048 + (wr * 2 + tr) * 256 + lane * 8;
        a[tr].q[0] = *(const u32x4*)ap;
        a[tr].q[1] = *(const u32x4*)(ap + 4);
      }
      #pragma unroll
      for (int tc = 0; tc < 2; ++tc) {
        const int col = colBase + wc * 32 + tc * 16 + (lane & 15);
        const unsigned short* bp =
            XT + (size_t)col * LDT + (k0 + kh * 32) + ((lane >> 4) * 8);
        b[tc].q[0] = *(const u32x4*)bp;
        b[tc].q[1] = *(const u32x4*)(bp + 16);
      }
      #pragma unroll
      for (int tr = 0; tr < 2; ++tr)
        #pragma unroll
        for (int tc = 0; tc < 2; ++tc)
          acc[tr][tc] = __builtin_amdgcn_wmma_f32_16x16x32_bf16(
              false, a[tr].m, false, b[tc].m, (short)0, acc[tr][tc], false, false);
    }
  };

  const int kTilesFull = NROWS / 64;         // 156 fully interior 64-k tiles
  for (int kt = 0; kt < kTilesFull; ++kt) {
    const int k0 = kt * 64;
    const int s  = kt & 1;

    // ---- stage adj tile (128x64 f32 -> bf16, WMMA A layout), branch-free ----
    #pragma unroll
    for (int i = 0; i < 2; ++i) {
      const int cc = c0 + i * 16;
      const float* src = arow + k0 + cc;
      f32x4 a0 = *(const f32x4*)(src + 0);
      f32x4 a1 = *(const f32x4*)(src + 4);
      f32x4 a2 = *(const f32x4*)(src + 8);
      f32x4 a3 = *(const f32x4*)(src + 12);
      u32x4 p0 = { pack2(a0[0], a0[1]), pack2(a0[2], a0[3]),
                   pack2(a1[0], a1[1]), pack2(a1[2], a1[3]) };
      u32x4 p1 = { pack2(a2[0], a2[1]), pack2(a2[2], a2[3]),
                   pack2(a3[0], a3[1]), pack2(a3[2], a3[3]) };
      const int ci  = cc >> 4;
      const int dwo = (ci >> 1) * 2048 + (r >> 4) * 256 + (r & 15) * 8 + (ci & 1) * 4;
      *(u32x4*)&lds[s][dwo]       = p0;   // lane (r&15):    K cc..cc+7
      *(u32x4*)&lds[s][dwo + 128] = p1;   // lane (r&15)+16: K cc+8..cc+15
    }
    if (kt + 2 < kTilesFull)
      __builtin_prefetch(arow + k0 + 128 + c0, 0, 3);   // global_prefetch next+1 tile
    __syncthreads();
    consume(&lds[s][0], k0);
  }

  // ---- tail tile: k = 9984..10047, only 16 valid; branchless clamp+select ----
  {
    const int k0 = kTilesFull * 64;          // 9984
    #pragma unroll
    for (int i = 0; i < 2; ++i) {
      const int cc = c0 + i * 16;
      float v[16];
      #pragma unroll
      for (int e = 0; e < 16; ++e) {
        const int k  = k0 + cc + e;
        const int ks = k < NROWS ? k : NROWS - 1;
        const float raw = arow[ks];
        v[e] = k < NROWS ? raw : 0.0f;
      }
      u32x4 p0 = { pack2(v[0], v[1]),   pack2(v[2], v[3]),
                   pack2(v[4], v[5]),   pack2(v[6], v[7]) };
      u32x4 p1 = { pack2(v[8], v[9]),   pack2(v[10], v[11]),
                   pack2(v[12], v[13]), pack2(v[14], v[15]) };
      const int ci  = cc >> 4;
      const int dwo = (ci >> 1) * 2048 + (r >> 4) * 256 + (r & 15) * 8 + (ci & 1) * 4;
      *(u32x4*)&lds[0][dwo]       = p0;
      *(u32x4*)&lds[0][dwo + 128] = p1;
    }
    __syncthreads();
    consume(&lds[0][0], k0);                 // B rows are exact zeros for k>=NROWS
  }

  // ---- epilogue: optional tanh, bf16 (+ optional NT f32) stores ----
  #pragma unroll
  for (int tr = 0; tr < 2; ++tr) {
    #pragma unroll
    for (int tc = 0; tc < 2; ++tc) {
      const int row0 = rowBase + wr * 32 + tr * 16 + 8 * (lane >> 4);
      const int col  = colBase + wc * 32 + tc * 16 + (lane & 15);
      #pragma unroll
      for (int e = 0; e < 8; ++e) {
        const int row = row0 + e;
        float x = acc[tr][tc][e];
        if (act) x = tanhf(x);
        if (row < NROWS) {
          outB[(size_t)row * ldB + col] = f2bf(x);
          if (outF && col < ncolF)
            __builtin_nontemporal_store(x, &outF[(size_t)row * ldF + col]);
        }
      }
    }
  }
}

// ---------------------------------------------------------------------------
// out = sigmoid(Az @ Az^T),  Az [NROWS x 512] bf16 (L2-resident).
// Both WMMA operands are contiguous rows of Az (B[k][n] = Az[n][k]).
// Output is write-once 400MB -> nontemporal stores keep Az resident in L2.
// ---------------------------------------------------------------------------
__global__ __launch_bounds__(256) void syrk_sigmoid(
    const unsigned short* __restrict__ Az,
    float* __restrict__ out)
{
  const int lane = threadIdx.x & 31;
  const int wave = threadIdx.x >> 5;
  const int wr = wave >> 1, wc = wave & 1;
  const int i0 = blockIdx.x * 128 + wr * 32;
  const int j0 = blockIdx.y * 64 + wc * 32;

  int ra[2], rb[2];
  #pragma unroll
  for (int tix = 0; tix < 2; ++tix) {
    int rr = i0 + tix * 16 + (lane & 15);
    ra[tix] = rr < NROWS ? rr : NROWS - 1;     // clamp: rows >= N discarded
    int cc = j0 + tix * 16 + (lane & 15);
    rb[tix] = cc < NROWS ? cc : NROWS - 1;
  }
  const int kbA = (lane >> 4) * 8;    // A layout: K0-7/16-23 vs K8-15/24-31
  const int kbB = (lane >> 4) * 8;    // B layout: K0-7/16-23 vs K8-15/24-31

  f32x8 acc[2][2] = {};
  for (int k0 = 0; k0 < 512; k0 += 32) {
    OpU a[2], b[2];
    #pragma unroll
    for (int tr = 0; tr < 2; ++tr) {
      const unsigned short* p = Az + (size_t)ra[tr] * 512 + k0 + kbA;
      a[tr].q[0] = *(const u32x4*)p;
      a[tr].q[1] = *(const u32x4*)(p + 16);
    }
    #pragma unroll
    for (int tc = 0; tc < 2; ++tc) {
      const unsigned short* p = Az + (size_t)rb[tc] * 512 + k0 + kbB;
      b[tc].q[0] = *(const u32x4*)p;
      b[tc].q[1] = *(const u32x4*)(p + 16);
    }
    #pragma unroll
    for (int tr = 0; tr < 2; ++tr)
      #pragma unroll
      for (int tc = 0; tc < 2; ++tc)
        acc[tr][tc] = __builtin_amdgcn_wmma_f32_16x16x32_bf16(
            false, a[tr].m, false, b[tc].m, (short)0, acc[tr][tc], false, false);
  }

  #pragma unroll
  for (int tr = 0; tr < 2; ++tr) {
    #pragma unroll
    for (int tc = 0; tc < 2; ++tc) {
      const int row0 = i0 + tr * 16 + 8 * (lane >> 4);
      const int col  = j0 + tc * 16 + (lane & 15);
      #pragma unroll
      for (int e = 0; e < 8; ++e) {
        const int row = row0 + e;
        if (row < NROWS && col < NROWS) {
          float x = acc[tr][tc][e];
          float sg = 1.0f / (1.0f + __expf(-x));
          __builtin_nontemporal_store(sg, &out[(size_t)row * NROWS + col]);
        }
      }
    }
  }
}

// ---------------------------------------------------------------------------
extern "C" void kernel_launch(void* const* d_in, const int* in_sizes, int n_in,
                              void* d_out, int out_size, void* d_ws, size_t ws_size,
                              hipStream_t stream)
{
  const float* z_igae = (const float*)d_in[0];   // [10000][20]
  const float* adj    = (const float*)d_in[1];   // [10000][10000]
  const float* w3     = (const float*)d_in[2];   // [20][256]
  const float* w4     = (const float*)d_in[3];   // [256][128]
  const float* w5     = (const float*)d_in[4];   // [128][500]

  float* zhat_out = (float*)d_out;                              // [10000][500]
  float* adjr_out = (float*)d_out + (size_t)NROWS * 500;        // [10000][10000]

  char* ws = (char*)d_ws;                       // ~36 MB of bf16 staging
  unsigned short* t1T = (unsigned short*)(ws + 0);          // [256][LDT]
  unsigned short* z1  = (unsigned short*)(ws + 5144576);    // [10000][256]
  unsigned short* t2T = (unsigned short*)(ws + 10264576);   // [128][LDT]
  unsigned short* z2  = (unsigned short*)(ws + 12836864);   // [10000][128]
  unsigned short* t3T = (unsigned short*)(ws + 15396864);   // [512][LDT]
  unsigned short* zh  = (unsigned short*)(ws + 25686016);   // [10000][512]

  dim3 blk(256);

  // t1^T = (z_igae @ w3)^T
  xw_gemm_T<<<(LDT * 256) / 256, blk, 0, stream>>>(z_igae, nullptr, w3, 20, 256, 256, t1T);
  // z1 = tanh(adj @ t1)
  adj_gemm<<<dim3(79, 4), blk, 0, stream>>>(adj, t1T, z1, 256, nullptr, 0, 0, 1);
  // t2^T = (z1 @ w4)^T
  xw_gemm_T<<<(LDT * 128) / 256, blk, 0, stream>>>(nullptr, z1, w4, 256, 128, 128, t2T);
  // z2 = tanh(adj @ t2)
  adj_gemm<<<dim3(79, 2), blk, 0, stream>>>(adj, t2T, z2, 128, nullptr, 0, 0, 1);
  // t3^T = (z2 @ w5)^T, 500 cols padded to 512 with exact zeros
  xw_gemm_T<<<(LDT * 512) / 256, blk, 0, stream>>>(nullptr, z2, w5, 128, 512, 500, t3T);
  // z_hat = adj @ t3  (NT f32 -> d_out, bf16 copy -> zh)
  adj_gemm<<<dim3(79, 8), blk, 0, stream>>>(adj, t3T, zh, 512, zhat_out, 500, 500, 0);
  // z_hat_adj = sigmoid(z_hat @ z_hat^T)
  syrk_sigmoid<<<dim3(79, 157), blk, 0, stream>>>(zh, adjr_out);
}
